// RGAC_6545530159350
// MI455X (gfx1250) — matrix operand order repaired and checked by reading, not compiled, not run on hardware
//
#include <hip/hip_runtime.h>

#define B_   32
#define C_   512
#define HW_  1024
#define OUT_ 256
static constexpr float EPS = 1e-5f;

typedef __attribute__((ext_vector_type(16))) __bf16 v16bf;
typedef __attribute__((ext_vector_type(8)))  __bf16 v8bf;
typedef __attribute__((ext_vector_type(8)))  float  v8f;

// ---------------------------------------------------------------------------
// K1a: convert x (f32) -> bf16 in natural [b][c][q] layout AND transposed
//      [b][q][c] layout (LDS-tiled so both global streams are coalesced).
// ---------------------------------------------------------------------------
__global__ void k_conv_x(const float* __restrict__ x,
                         __bf16* __restrict__ xh,    // [B][C][HW]
                         __bf16* __restrict__ xhT) { // [B][HW][C]
    __shared__ __bf16 tile[32][33];
    const int b  = blockIdx.z;
    const int c0 = blockIdx.y * 32;
    const int q0 = blockIdx.x * 32;
    const int tx = threadIdx.x;   // 0..31 (q within tile)
    const int ty = threadIdx.y;   // 0..7
#pragma unroll
    for (int i = 0; i < 4; ++i) {
        const int cl = ty + i * 8;
        const size_t idx = ((size_t)b * C_ + (c0 + cl)) * HW_ + (q0 + tx);
        const __bf16 h = (__bf16)x[idx];
        xh[idx] = h;
        tile[tx][cl] = h;           // tile[q_local][c_local]
    }
    __syncthreads();
#pragma unroll
    for (int i = 0; i < 4; ++i) {
        const int ql = ty + i * 8;
        xhT[((size_t)b * HW_ + (q0 + ql)) * C_ + (c0 + tx)] = tile[ql][tx];
    }
}

// ---------------------------------------------------------------------------
// K1b: convert w1, w2 (256x1024) and w3 (512x512) to bf16 (all 262144 elems).
// ---------------------------------------------------------------------------
__global__ void k_conv_w(const float* __restrict__ w1, const float* __restrict__ w2,
                         const float* __restrict__ w3,
                         __bf16* __restrict__ w1h, __bf16* __restrict__ w2h,
                         __bf16* __restrict__ w3h) {
    const int i   = blockIdx.x * blockDim.x + threadIdx.x;
    const int seg = i >> 18;          // 262144 = 2^18
    const int j   = i & 262143;
    if      (seg == 0) w1h[j] = (__bf16)w1[j];
    else if (seg == 1) w2h[j] = (__bf16)w2[j];
    else               w3h[j] = (__bf16)w3[j];
}

// K1c: zero the xe atomic accumulator (must be re-zeroed every call).
__global__ void k_zero(float* __restrict__ p) {
    p[blockIdx.x * blockDim.x + threadIdx.x] = 0.f;
}

// ---------------------------------------------------------------------------
// K2: theta / phi GEMM via WMMA bf16.
//   D[m=c][n=o] = sum_q xh[b][c][q] * W[o][q]   (A = x row-contig, B = W row-contig)
//   Fused +bias, BN(eval), ReLU; store bf16 transposed: thT/phT[b][c][o].
//   One wave computes a 16(M=c) x 64(N=o) macro-tile (A frag reused 4x).
// ---------------------------------------------------------------------------
__global__ void __launch_bounds__(32) k_gemm_theta_phi(
    const __bf16* __restrict__ xh, const __bf16* __restrict__ w1h,
    const __bf16* __restrict__ w2h,
    const float* __restrict__ b1, const float* __restrict__ g1,
    const float* __restrict__ be1, const float* __restrict__ m1,
    const float* __restrict__ v1,
    const float* __restrict__ b2, const float* __restrict__ g2,
    const float* __restrict__ be2, const float* __restrict__ m2,
    const float* __restrict__ v2,
    __bf16* __restrict__ thT, __bf16* __restrict__ phT) {
    const int lane  = threadIdx.x & 31;
    const int otile = blockIdx.x * 64;        // 4 blocks  -> OUT
    const int ctile = blockIdx.y * 16;        // 32 blocks -> C
    const int b     = blockIdx.z >> 1;
    const bool second = blockIdx.z & 1;

    const __bf16* __restrict__ W = second ? w2h : w1h;
    const float* bb = second ? b2 : b1;   const float* gg = second ? g2 : g1;
    const float* be = second ? be2 : be1; const float* mm = second ? m2 : m1;
    const float* vv = second ? v2 : v1;
    __bf16* __restrict__ outp = second ? phT : thT;

    const int half = lane >> 4;    // K-half selector per WMMA layout
    const int lan  = lane & 15;    // M row (A) / N col (B)

    // A: row c = ctile+lan, K run {kh..kh+15}, kh = half*16
    const __bf16* aptr = xh + ((size_t)b * C_ + (ctile + lan)) * HW_ + half * 16;
    // B: col o, K run {kbase..kbase+15}, kbase = half*16 (32B contiguous)
    const __bf16* bptr[4];
#pragma unroll
    for (int s = 0; s < 4; ++s)
        bptr[s] = W + (size_t)(otile + s * 16 + lan) * HW_ + half * 16;

    const v8f vzero = {0.f,0.f,0.f,0.f,0.f,0.f,0.f,0.f};
    v8f acc[4] = {vzero, vzero, vzero, vzero};

    for (int k = 0; k < HW_; k += 32) {
        const v8bf a0 = *(const v8bf*)(aptr + k);
        const v8bf a1 = *(const v8bf*)(aptr + k + 8);
        const v16bf af = __builtin_shufflevector(a0, a1,
            0,1,2,3,4,5,6,7,8,9,10,11,12,13,14,15);
#pragma unroll
        for (int s = 0; s < 4; ++s) {
            const v16bf bf = *(const v16bf*)(bptr[s] + k);
            acc[s] = __builtin_amdgcn_wmma_f32_16x16x32_bf16(
                false, af, false, bf, (short)0, acc[s], false, false);
        }
    }

    // Fused bias + BN(eval) + ReLU, store transposed [c][o] as bf16.
#pragma unroll
    for (int s = 0; s < 4; ++s) {
        const int oo = otile + s * 16 + lan;               // N column
        const float alpha = gg[oo] * rsqrtf(vv[oo] + EPS);
        const float cb    = (bb[oo] - mm[oo]) * alpha + be[oo];
#pragma unroll
        for (int r = 0; r < 8; ++r) {
            const int cc = ctile + r + half * 8;           // M row
            float val = acc[s][r] * alpha + cb;
            val = val > 0.f ? val : 0.f;
            outp[((size_t)b * C_ + cc) * OUT_ + oo] = (__bf16)val;
        }
    }
}

// ---------------------------------------------------------------------------
// K3: xe GEMM via WMMA bf16.
//   D[m=o][n=hw] = sum_c w3[o][c] * xT[b][hw][c]
//   Fused +b3, BN, ReLU; shfl-reduce 16 lanes sharing the same o-row, then
//   atomicAdd into xesum[b][o] (the HW mean, divided by 1024 later).
// ---------------------------------------------------------------------------
__global__ void __launch_bounds__(32) k_gemm_xe(
    const __bf16* __restrict__ xhT, const __bf16* __restrict__ w3h,
    const float* __restrict__ b3, const float* __restrict__ g3,
    const float* __restrict__ be3, const float* __restrict__ m3,
    const float* __restrict__ v3,
    float* __restrict__ xesum) {
    const int lane   = threadIdx.x & 31;
    const int otile  = blockIdx.x * 16;   // 32 blocks -> C (o dim = 512)
    const int hwtile = blockIdx.y * 64;   // 16 blocks -> HW
    const int b      = blockIdx.z;

    const int half = lane >> 4;
    const int lan  = lane & 15;

    const __bf16* aptr = w3h + (size_t)(otile + lan) * C_ + half * 16;
    const __bf16* bptr[4];
#pragma unroll
    for (int s = 0; s < 4; ++s)
        bptr[s] = xhT + ((size_t)b * HW_ + (hwtile + s * 16 + lan)) * C_ + half * 16;

    const v8f vzero = {0.f,0.f,0.f,0.f,0.f,0.f,0.f,0.f};
    v8f acc[4] = {vzero, vzero, vzero, vzero};

    for (int k = 0; k < C_; k += 32) {
        const v8bf a0 = *(const v8bf*)(aptr + k);
        const v8bf a1 = *(const v8bf*)(aptr + k + 8);
        const v16bf af = __builtin_shufflevector(a0, a1,
            0,1,2,3,4,5,6,7,8,9,10,11,12,13,14,15);
#pragma unroll
        for (int s = 0; s < 4; ++s) {
            const v16bf bf = *(const v16bf*)(bptr[s] + k);
            acc[s] = __builtin_amdgcn_wmma_f32_16x16x32_bf16(
                false, af, false, bf, (short)0, acc[s], false, false);
        }
    }

#pragma unroll
    for (int s = 0; s < 4; ++s) {
#pragma unroll
        for (int r = 0; r < 8; ++r) {
            const int oo = otile + r + half * 8;           // M row (o)
            const float alpha = g3[oo] * rsqrtf(v3[oo] + EPS);
            const float cb    = (b3[oo] - m3[oo]) * alpha + be3[oo];
            float val = acc[s][r] * alpha + cb;
            val = val > 0.f ? val : 0.f;
            // 16 lanes of this half share oo -> butterfly reduce over N.
            val += __shfl_xor(val, 1, 32);
            val += __shfl_xor(val, 2, 32);
            val += __shfl_xor(val, 4, 32);
            val += __shfl_xor(val, 8, 32);
            if (lan == 0) atomicAdd(xesum + (size_t)b * C_ + oo, val);
        }
    }
}

// ---------------------------------------------------------------------------
// K4a: collapse the aff GEMM, stage 1:
//   u[b,k] = sum_c w4[1+c]   * phi[b,k,c]   (phi[b,k,c]   = phT[b][c][k])
//   v[b,k] = sum_c w4[1+C+c] * theta[b,k,c] (theta[b,k,c] = thT[b][c][k])
// ---------------------------------------------------------------------------
__global__ void k_reduce_uv(const __bf16* __restrict__ thT,
                            const __bf16* __restrict__ phT,
                            const float* __restrict__ w4,
                            float* __restrict__ u, float* __restrict__ v) {
    const int t = blockIdx.x * blockDim.x + threadIdx.x; // [0, B*OUT)
    const int b = t / OUT_, k = t % OUT_;
    const __bf16* pp = phT + (size_t)b * C_ * OUT_ + k;
    const __bf16* tt = thT + (size_t)b * C_ * OUT_ + k;
    float su = 0.f, sv = 0.f;
    for (int c = 0; c < C_; ++c) {
        su += w4[1 + c]      * (float)pp[(size_t)c * OUT_];
        sv += w4[1 + C_ + c] * (float)tt[(size_t)c * OUT_];
    }
    u[t] = su; v[t] = sv;
}

// K4b: stage 2:
//   ra[b,p] = sum_k u[b,k]*theta[b,k,p] = sum_k u[b,k]*thT[b][p][k]
//   rb[b,p] = sum_k v[b,k]*phi[b,k,p]   = sum_k v[b,k]*phT[b][p][k]
__global__ void k_reduce_rarb(const __bf16* __restrict__ thT,
                              const __bf16* __restrict__ phT,
                              const float* __restrict__ u, const float* __restrict__ v,
                              float* __restrict__ ra, float* __restrict__ rb) {
    const int t = blockIdx.x * blockDim.x + threadIdx.x; // [0, B*C)
    const int b = t / C_, p = t % C_;
    const __bf16* tt = thT + ((size_t)b * C_ + p) * OUT_;
    const __bf16* pp = phT + ((size_t)b * C_ + p) * OUT_;
    const float* ub = u + (size_t)b * OUT_;
    const float* vb = v + (size_t)b * OUT_;
    float sa = 0.f, sb = 0.f;
    for (int k = 0; k < OUT_; ++k) {
        sa += ub[k] * (float)tt[k];
        sb += vb[k] * (float)pp[k];
    }
    ra[t] = sa; rb[t] = sb;
}

// K5: logits + sigmoid -> a[b][c]
__global__ void k_act(const float* __restrict__ xesum,
                      const float* __restrict__ ra, const float* __restrict__ rb,
                      const float* __restrict__ w4, const float* __restrict__ b4,
                      float* __restrict__ a) {
    const int t = blockIdx.x * blockDim.x + threadIdx.x; // [0, B*C)
    const float xe = xesum[t] * (1.f / (float)HW_);
    const float logit = w4[0] * xe + ra[t] + rb[t] + b4[0];
    a[t] = 1.f / (1.f + __expf(-logit));
}

// K6: out = x * a[b,c]  (float4, HW divisible by 4 -> one scale per vec4)
__global__ void k_final(const float* __restrict__ x, const float* __restrict__ a,
                        float* __restrict__ out) {
    const size_t i4 = (size_t)blockIdx.x * blockDim.x + threadIdx.x;
    const float s = a[i4 >> 8];                 // (i4*4)/HW_, HW_/4 = 256
    const float4 xv = ((const float4*)x)[i4];
    ((float4*)out)[i4] = make_float4(xv.x * s, xv.y * s, xv.z * s, xv.w * s);
}

// ---------------------------------------------------------------------------
extern "C" void kernel_launch(void* const* d_in, const int* in_sizes, int n_in,
                              void* d_out, int out_size, void* d_ws, size_t ws_size,
                              hipStream_t stream) {
    const float* x   = (const float*)d_in[0];
    const float* w1  = (const float*)d_in[1];
    const float* b1  = (const float*)d_in[2];
    const float* g1  = (const float*)d_in[3];
    const float* be1 = (const float*)d_in[4];
    const float* m1  = (const float*)d_in[5];
    const float* v1  = (const float*)d_in[6];
    const float* w2  = (const float*)d_in[7];
    const float* b2  = (const float*)d_in[8];
    const float* g2  = (const float*)d_in[9];
    const float* be2 = (const float*)d_in[10];
    const float* m2  = (const float*)d_in[11];
    const float* v2  = (const float*)d_in[12];
    const float* w3  = (const float*)d_in[13];
    const float* b3  = (const float*)d_in[14];
    const float* g3  = (const float*)d_in[15];
    const float* be3 = (const float*)d_in[16];
    const float* m3  = (const float*)d_in[17];
    const float* v3  = (const float*)d_in[18];
    const float* w4  = (const float*)d_in[19];
    const float* b4  = (const float*)d_in[20];
    float* out = (float*)d_out;

    // Workspace layout
    char* ws = (char*)d_ws;
    constexpr size_t SZ_XH  = (size_t)B_ * C_ * HW_ * 2;   // 32 MB
    constexpr size_t SZ_THT = (size_t)B_ * C_ * OUT_ * 2;  //  8 MB
    constexpr size_t SZ_W   = (size_t)256 * 1024 * 2;      // 512 KB (also w3)
    size_t off = 0;
    __bf16* xh   = (__bf16*)(ws + off); off += SZ_XH;
    __bf16* xhT  = (__bf16*)(ws + off); off += SZ_XH;
    __bf16* thT  = (__bf16*)(ws + off); off += SZ_THT;
    __bf16* phT  = (__bf16*)(ws + off); off += SZ_THT;
    __bf16* w1h  = (__bf16*)(ws + off); off += SZ_W;
    __bf16* w2h  = (__bf16*)(ws + off); off += SZ_W;
    __bf16* w3h  = (__bf16*)(ws + off); off += SZ_W;
    float*  xes  = (float*)(ws + off);  off += (size_t)B_ * C_ * 4;
    float*  u    = (float*)(ws + off);  off += (size_t)B_ * OUT_ * 4;
    float*  v    = (float*)(ws + off);  off += (size_t)B_ * OUT_ * 4;
    float*  ra   = (float*)(ws + off);  off += (size_t)B_ * C_ * 4;
    float*  rb   = (float*)(ws + off);  off += (size_t)B_ * C_ * 4;
    float*  av   = (float*)(ws + off);  off += (size_t)B_ * C_ * 4;

    // 1) conversions / transpose / zero accumulator
    k_conv_x<<<dim3(HW_ / 32, C_ / 32, B_), dim3(32, 8, 1), 0, stream>>>(x, xh, xhT);
    k_conv_w<<<dim3(3 * 262144 / 256), dim3(256), 0, stream>>>(w1, w2, w3, w1h, w2h, w3h);
    k_zero<<<dim3(B_ * C_ / 256), dim3(256), 0, stream>>>(xes);

    // 2) theta + phi WMMA GEMMs (fused bias+BN+ReLU, transposed bf16 out)
    k_gemm_theta_phi<<<dim3(OUT_ / 64, C_ / 16, B_ * 2), dim3(32), 0, stream>>>(
        xh, w1h, w2h, b1, g1, be1, m1, v1, b2, g2, be2, m2, v2, thT, phT);

    // 3) xe WMMA GEMM (fused bias+BN+ReLU + HW-sum reduction)
    k_gemm_xe<<<dim3(C_ / 16, HW_ / 64, B_), dim3(32), 0, stream>>>(
        xhT, w3h, b3, g3, be3, m3, v3, xes);

    // 4) collapsed aff contractions (two mat-vec passes instead of 512x512 GEMM)
    k_reduce_uv<<<dim3(B_ * OUT_ / 256), dim3(256), 0, stream>>>(thT, phT, w4, u, v);
    k_reduce_rarb<<<dim3(B_ * C_ / 256), dim3(256), 0, stream>>>(thT, phT, u, v, ra, rb);

    // 5) sigmoid gate, then broadcast multiply
    k_act<<<dim3(B_ * C_ / 256), dim3(256), 0, stream>>>(xes, ra, rb, w4, b4, av);
    k_final<<<dim3((size_t)B_ * C_ * HW_ / 4 / 256), dim3(256), 0, stream>>>(x, av, out);

    (void)in_sizes; (void)n_in; (void)out_size; (void)ws_size;
}
